// PatternEncoder_5703716569751
// MI455X (gfx1250) — matrix operand order repaired and checked by reading, not compiled, not run on hardware
//
#include <hip/hip_runtime.h>
#include <cstdint>
#include <cstddef>

typedef __attribute__((ext_vector_type(16))) _Float16     v16h;
typedef __attribute__((ext_vector_type(8)))  float        v8f;
typedef __attribute__((ext_vector_type(4)))  unsigned int v4u;
typedef __attribute__((ext_vector_type(4)))  float        v4f;

static constexpr int N_   = 50000;   // nodes
static constexpr int E_   = 300000;  // edges
static constexpr int B_   = 64;      // graphs
static constexpr int EXT_ = 8;       // node feature dim
static constexpr int H_   = 512;     // hidden
static constexpr int H3_  = 1536;    // 3*H
static constexpr int D_   = 3096;    // combined readout dim
static constexpr int KP1_ = 32;      // EXT padded to K%32
static constexpr int KHD_ = 3104;    // 3098/3102 padded to K%32

union HV { v4u q[2]; v16h h; };

__device__ __forceinline__ void atomAddF(float* p, float v) {
  (void)__hip_atomic_fetch_add(p, v, __ATOMIC_RELAXED, __HIP_MEMORY_SCOPE_AGENT);
}

// ---------------------------------------------------------------- utility
__global__ __launch_bounds__(256) void k_fill_zero(float* __restrict__ p, long long n) {
  long long i = (long long)blockIdx.x * blockDim.x + threadIdx.x;
  if (i < n) p[i] = 0.f;
}

__global__ __launch_bounds__(256) void k_fill_zero_i(int* __restrict__ p, int n) {
  int i = blockIdx.x * blockDim.x + threadIdx.x;
  if (i < n) p[i] = 0;
}

__global__ __launch_bounds__(256) void k_deg_count(const int* __restrict__ dst,
                                                   float* __restrict__ deg, int e) {
  int i = blockIdx.x * blockDim.x + threadIdx.x;
  if (i < e) atomAddF(&deg[dst[i]], 1.0f);
}

__global__ __launch_bounds__(256) void k_batch_count(const int* __restrict__ batch,
                                                     int* __restrict__ cnts, int n) {
  int i = blockIdx.x * blockDim.x + threadIdx.x;
  if (i < n) atomicAdd(&cnts[batch[i]], 1);
}

__global__ void k_deg_finish(const float* __restrict__ deg, float* __restrict__ dis,
                             float* __restrict__ selfn, int n) {
  int i = blockIdx.x * blockDim.x + threadIdx.x;
  if (i >= n) return;
  float d = deg[i] + 1.0f;        // self loop
  dis[i] = rsqrtf(d);
  selfn[i] = 1.0f / d;
}

__global__ void k_scan_starts(const int* __restrict__ cnts, int* __restrict__ starts, int b) {
  if (threadIdx.x == 0 && blockIdx.x == 0) {
    int a = 0;
    for (int g = 0; g < b; ++g) { starts[g] = a; a += cnts[g]; }
  }
}

// f32 [rows,cs] -> f16 [rows,cd] zero-padded columns
__global__ __launch_bounds__(256) void k_cvt_pad(const float* __restrict__ src,
                                                 _Float16* __restrict__ dst,
                                                 long long total, int cs, int cd) {
  long long i = (long long)blockIdx.x * blockDim.x + threadIdx.x;
  if (i >= total) return;
  int r = (int)(i / cd), c = (int)(i % cd);
  dst[i] = (c < cs) ? (_Float16)src[(long long)r * cs + c] : (_Float16)0.f;
}

// W f32 [K,Ncols] -> Wt f16 [Ncols,Kpad] (transposed, K zero-padded)
__global__ __launch_bounds__(256) void k_cvt_transpose(const float* __restrict__ W,
                                                       _Float16* __restrict__ Wt,
                                                       long long total, int K, int Ncols, int Kpad) {
  long long i = (long long)blockIdx.x * blockDim.x + threadIdx.x;
  if (i >= total) return;
  int col = (int)(i / Kpad), k = (int)(i % Kpad);
  Wt[i] = (k < K) ? (_Float16)W[(long long)k * Ncols + col] : (_Float16)0.f;
}

// ---------------------------------------------------------------- WMMA GEMM
// C[M,Nc] f32 = A[M,K] f16 (row major) @ Bt[Nc,K] f16 (B transposed).
// Requires M%16==0, K%32==0, Nc%64==0. One wave -> 16x64 tile, 4 WMMAs/K-step.
__global__ __launch_bounds__(256) void k_gemm_wmma(const _Float16* __restrict__ A,
                                                   const _Float16* __restrict__ Bt,
                                                   float* __restrict__ C,
                                                   int M, int K, int Nc) {
  const int lane = threadIdx.x & 31;
  const int wave = blockIdx.x * (blockDim.x >> 5) + (threadIdx.x >> 5);
  const int tilesN = Nc >> 6;
  const int tilesM = M >> 4;
  if (wave >= tilesM * tilesN) return;          // wave-uniform: EXEC stays all-1s
  const int mt = wave / tilesN;
  const int nt = wave % tilesN;
  const int half = lane >> 4;                   // 0 | 1
  const int sub  = lane & 15;
  const int arow = (mt << 4) + sub;
  const int n0   = nt << 6;

  v8f acc0 = {}, acc1 = {}, acc2 = {}, acc3 = {};

  // A frag: lane holds row `arow`; half0 -> K 0..15 contiguous, half1 -> K 16..31
  const v4u* Ap = (const v4u*)(A + (size_t)arow * K);
  // Bt frag: lane holds column n; half0 -> K 0..15 contiguous, half1 -> K 16..31
  const v4u* B0 = (const v4u*)(Bt + (size_t)(n0 +  0 + sub) * K);
  const v4u* B1 = (const v4u*)(Bt + (size_t)(n0 + 16 + sub) * K);
  const v4u* B2 = (const v4u*)(Bt + (size_t)(n0 + 32 + sub) * K);
  const v4u* B3 = (const v4u*)(Bt + (size_t)(n0 + 48 + sub) * K);
  const int aо0 = half * 2, ao1 = half * 2 + 1;
  const int bo0 = half * 2, bo1 = half * 2 + 1;

  for (int k0 = 0; k0 < K; k0 += 32) {
    HV a;  a.q[0]  = Ap[aо0]; a.q[1]  = Ap[ao1];
    HV b0; b0.q[0] = B0[bo0]; b0.q[1] = B0[bo1];
    HV b1; b1.q[0] = B1[bo0]; b1.q[1] = B1[bo1];
    HV b2; b2.q[0] = B2[bo0]; b2.q[1] = B2[bo1];
    HV b3; b3.q[0] = B3[bo0]; b3.q[1] = B3[bo1];
    acc0 = __builtin_amdgcn_wmma_f32_16x16x32_f16(false, a.h, false, b0.h, (short)0, acc0, false, false);
    acc1 = __builtin_amdgcn_wmma_f32_16x16x32_f16(false, a.h, false, b1.h, (short)0, acc1, false, false);
    acc2 = __builtin_amdgcn_wmma_f32_16x16x32_f16(false, a.h, false, b2.h, (short)0, acc2, false, false);
    acc3 = __builtin_amdgcn_wmma_f32_16x16x32_f16(false, a.h, false, b3.h, (short)0, acc3, false, false);
    Ap += 4; B0 += 4; B1 += 4; B2 += 4; B3 += 4;
  }

  // D layout: lane col = sub (+tile), VGPR v -> row v + half*8
  float* Cp = C + (size_t)((mt << 4) + half * 8) * Nc + n0 + sub;
  #pragma unroll
  for (int v = 0; v < 8; ++v) {
    Cp[(size_t)v * Nc +  0] = acc0[v];
    Cp[(size_t)v * Nc + 16] = acc1[v];
    Cp[(size_t)v * Nc + 32] = acc2[v];
    Cp[(size_t)v * Nc + 48] = acc3[v];
  }
}

// ---------------------------------------------------------------- GCN pieces
// agg[dst] += t[src] * dis[src]   (thread = edge x 4 channels)
__global__ __launch_bounds__(256) void k_edge_agg(const int* __restrict__ src,
                                                  const int* __restrict__ dst,
                                                  const float* __restrict__ t,
                                                  const float* __restrict__ dis,
                                                  float* __restrict__ agg,
                                                  long long total) {
  long long gid = (long long)blockIdx.x * blockDim.x + threadIdx.x;
  if (gid >= total) return;
  int e  = (int)(gid >> 7);          // H/4 = 128 chunks per edge
  int c4 = (int)(gid & 127) << 2;
  int s = src[e], d = dst[e];
  float ds = dis[s];
  v4f v = *(const v4f*)(t + (size_t)s * H_ + c4);
  float* ap = agg + (size_t)d * H_ + c4;
  atomAddF(ap + 0, v.x * ds);
  atomAddF(ap + 1, v.y * ds);
  atomAddF(ap + 2, v.z * ds);
  atomAddF(ap + 3, v.w * ds);
}

// h = relu(agg*dis + t*selfn + b) -> f16
__global__ __launch_bounds__(256) void k_gcn_finish(const float* __restrict__ t,
                                                    const float* __restrict__ agg,
                                                    const float* __restrict__ dis,
                                                    const float* __restrict__ selfn,
                                                    const float* __restrict__ b,
                                                    _Float16* __restrict__ hout,
                                                    long long total) {
  long long i = (long long)blockIdx.x * blockDim.x + threadIdx.x;
  if (i >= total) return;
  int node = (int)(i >> 9);
  int c    = (int)(i & 511);
  float v = agg[i] * dis[node] + t[i] * selfn[node] + b[c];
  hout[i] = (_Float16)(v > 0.f ? v : 0.f);
}

// per-graph [mean || sum] pool into comb[:, off .. off+1024)
__global__ __launch_bounds__(256) void k_readout(const _Float16* __restrict__ h,
                                                 const int* __restrict__ starts,
                                                 const int* __restrict__ cnts,
                                                 float* __restrict__ comb, int off) {
  int g = blockIdx.x;
  int c = blockIdx.y * 256 + threadIdx.x;     // gridDim.y == 2 -> H=512
  int s = starts[g], n = cnts[g];
  float sum = 0.f;
  for (int i = 0; i < n; ++i) sum += (float)h[(size_t)(s + i) * H_ + c];
  comb[(size_t)g * D_ + off + c]        = sum / (float)n;
  comb[(size_t)g * D_ + off + 512 + c]  = sum;
}

// ---------------------------------------------------------------- face (two smallest ||pos||)
__device__ __forceinline__ void ins2(float v, int i, float& v0, int& i0, float& v1, int& i1) {
  if (v < v0 || (v == v0 && i < i0)) { v1 = v0; i1 = i0; v0 = v; i0 = i; }
  else if (v < v1 || (v == v1 && i < i1)) { v1 = v; i1 = i; }
}

__global__ __launch_bounds__(256) void k_face(const float* __restrict__ x,
                                              const int* __restrict__ starts,
                                              const int* __restrict__ cnts,
                                              float* __restrict__ comb) {
  __shared__ float sv0[256], sv1[256];
  __shared__ int   si0[256], si1[256];
  int g = blockIdx.x, tid = threadIdx.x;
  int s = starts[g], c = cnts[g];
  float v0 = 3.0e38f, v1 = 3.0e38f;
  int   i0 = 0x7fffffff, i1 = 0x7fffffff;
  for (int n = s + tid; n < s + c; n += 256) {
    float dx = x[(size_t)n * EXT_ + 0];
    float dy = x[(size_t)n * EXT_ + 1];
    ins2(dx * dx + dy * dy, n, v0, i0, v1, i1);
  }
  sv0[tid] = v0; si0[tid] = i0; sv1[tid] = v1; si1[tid] = i1;
  __syncthreads();
  for (int off = 128; off > 0; off >>= 1) {
    if (tid < off) {
      float a0 = sv0[tid], a1 = sv1[tid];
      int   j0 = si0[tid], j1 = si1[tid];
      ins2(sv0[tid + off], si0[tid + off], a0, j0, a1, j1);
      ins2(sv1[tid + off], si1[tid + off], a0, j0, a1, j1);
      sv0[tid] = a0; si0[tid] = j0; sv1[tid] = a1; si1[tid] = j1;
    }
    __syncthreads();
  }
  if (tid < 24) {
    int j0 = si0[0], j1 = si1[0];
    float v;
    if (tid < 8)       v = (j0 != 0x7fffffff) ? x[(size_t)j0 * EXT_ + tid]       : 0.f;
    else if (tid < 16) v = (j1 != 0x7fffffff) ? x[(size_t)j1 * EXT_ + (tid - 8)] : 0.f;
    else               v = 0.f;
    comb[(size_t)g * D_ + tid] = v;
  }
}

// ---------------------------------------------------------------- heads
// dst f16 [B, KHD_] = [p1(n1) | p2(n2) | comb(D_) | zeros]
__global__ __launch_bounds__(256) void k_build_head_in(_Float16* __restrict__ dst,
                                                       const float* __restrict__ p1, int n1,
                                                       const float* __restrict__ p2, int n2,
                                                       const float* __restrict__ comb) {
  int gid = blockIdx.x * blockDim.x + threadIdx.x;
  if (gid >= B_ * KHD_) return;
  int g = gid / KHD_, c = gid % KHD_;
  float v;
  if (c < n1)                 v = p1[g * n1 + c];
  else if (c < n1 + n2)       v = p2[g * n2 + (c - n1)];
  else if (c < n1 + n2 + D_)  v = comb[(size_t)g * D_ + (c - n1 - n2)];
  else                        v = 0.f;
  dst[gid] = (_Float16)v;
}

__global__ __launch_bounds__(256) void k_bias_relu_cvt(const float* __restrict__ a,
                                                       const float* __restrict__ b,
                                                       _Float16* __restrict__ o,
                                                       long long total, int cols) {
  long long i = (long long)blockIdx.x * blockDim.x + threadIdx.x;
  if (i >= total) return;
  int c = (int)(i % cols);
  float v = a[i] + b[c];
  o[i] = (_Float16)(v > 0.f ? v : 0.f);
}

// out[g,o] = bo[o] + sum_k relu(a2[g,k]+b2[k]) * w[k,o]
__global__ __launch_bounds__(256) void k_proj(const float* __restrict__ a2,
                                              const float* __restrict__ b2,
                                              const float* __restrict__ w,
                                              const float* __restrict__ bo,
                                              float* __restrict__ out, int K, int nout) {
  int t = blockIdx.x * blockDim.x + threadIdx.x;
  if (t >= B_ * nout) return;
  int g = t / nout, o = t % nout;
  float s = bo[o];
  for (int k = 0; k < K; ++k) {
    float v = a2[(size_t)g * K + k] + b2[k];
    v = v > 0.f ? v : 0.f;
    s += v * w[(size_t)k * nout + o];
  }
  out[t] = s;
}

__global__ __launch_bounds__(256) void k_assemble(const float* __restrict__ pos,
                                                  const float* __restrict__ sc,
                                                  const float* __restrict__ rot,
                                                  const float* __restrict__ vec,
                                                  float* __restrict__ out) {
  int t = blockIdx.x * blockDim.x + threadIdx.x;
  if (t >= B_ * 8) return;
  int g = t >> 3, j = t & 7;
  float v;
  if (j < 2)      v = pos[g * 2 + j];
  else if (j == 2) v = sc[g];
  else if (j == 3) v = rot[g];
  else             v = vec[g * 4 + (j - 4)];
  out[t] = v;
}

// ---------------------------------------------------------------- host
static inline int NB(long long n) { return (int)((n + 255) / 256); }

extern "C" void kernel_launch(void* const* d_in, const int* in_sizes, int n_in,
                              void* d_out, int out_size, void* d_ws, size_t ws_size,
                              hipStream_t stream) {
  const float* x     = (const float*)d_in[0];
  const int*   ei    = (const int*)d_in[1];
  const int*   srcI  = ei;
  const int*   dstI  = ei + E_;
  const int*   batch = (const int*)d_in[2];
  const float* tpos  = (const float*)d_in[3];
  const float *W1 = (const float*)d_in[4],  *b1 = (const float*)d_in[5];
  const float *W2 = (const float*)d_in[6],  *b2 = (const float*)d_in[7];
  const float *W3 = (const float*)d_in[8],  *b3 = (const float*)d_in[9];
  const float *ph1w = (const float*)d_in[10], *ph1b = (const float*)d_in[11];
  const float *ph2w = (const float*)d_in[12], *ph2b = (const float*)d_in[13];
  const float *posw = (const float*)d_in[14], *posb_ = (const float*)d_in[15];
  const float *vh1w = (const float*)d_in[16], *vh1b = (const float*)d_in[17];
  const float *vh2w = (const float*)d_in[18], *vh2b = (const float*)d_in[19];
  const float *vecw = (const float*)d_in[20], *vecb_ = (const float*)d_in[21];
  const float *sh1w = (const float*)d_in[22], *sh1b = (const float*)d_in[23];
  const float *sh2w = (const float*)d_in[24], *sh2b = (const float*)d_in[25];
  const float *scw  = (const float*)d_in[26], *scb_ = (const float*)d_in[27];
  const float *rh1w = (const float*)d_in[28], *rh1b = (const float*)d_in[29];
  const float *rh2w = (const float*)d_in[30], *rh2b = (const float*)d_in[31];
  const float *rotw = (const float*)d_in[32], *rotb_ = (const float*)d_in[33];

  // bump allocator on workspace
  char* base = (char*)d_ws;
  size_t off = 0;
  auto alloc = [&](size_t bytes) -> void* {
    void* r = base + off;
    off += (bytes + 255) & ~(size_t)255;
    return r;
  };

  float*    deg    = (float*)alloc((size_t)N_ * 4);
  float*    dis    = (float*)alloc((size_t)N_ * 4);
  float*    selfn  = (float*)alloc((size_t)N_ * 4);
  int*      cnts   = (int*)alloc(B_ * 4);
  int*      starts = (int*)alloc(B_ * 4);
  _Float16* xh     = (_Float16*)alloc((size_t)N_ * KP1_ * 2);
  _Float16* hA     = (_Float16*)alloc((size_t)N_ * H_ * 2);
  _Float16* hB     = (_Float16*)alloc((size_t)N_ * H_ * 2);
  float*    tbuf   = (float*)alloc((size_t)N_ * H_ * 4);
  float*    agg    = (float*)alloc((size_t)N_ * H_ * 4);
  float*    comb   = (float*)alloc((size_t)B_ * D_ * 4);
  _Float16* hin    = (_Float16*)alloc((size_t)B_ * KHD_ * 2);
  float*    a1     = (float*)alloc((size_t)B_ * H3_ * 4);
  _Float16* a1h    = (_Float16*)alloc((size_t)B_ * H3_ * 2);
  float*    a2     = (float*)alloc((size_t)B_ * H3_ * 4);
  float*    posb   = (float*)alloc(B_ * 2 * 4);
  float*    vecb   = (float*)alloc(B_ * 4 * 4);
  float*    scb    = (float*)alloc(B_ * 4);
  float*    rotb   = (float*)alloc(B_ * 4);
  _Float16* W1t    = (_Float16*)alloc((size_t)H_ * KP1_ * 2);
  _Float16* W2t    = (_Float16*)alloc((size_t)H_ * H_ * 2);
  _Float16* W3t    = (_Float16*)alloc((size_t)H_ * H_ * 2);
  _Float16* ph1t   = (_Float16*)alloc((size_t)H3_ * KHD_ * 2);
  _Float16* ph2t   = (_Float16*)alloc((size_t)H3_ * H3_ * 2);
  _Float16* vh1t   = (_Float16*)alloc((size_t)H_ * KHD_ * 2);
  _Float16* vh2t   = (_Float16*)alloc((size_t)H_ * H_ * 2);
  _Float16* sh1t   = (_Float16*)alloc((size_t)H_ * KHD_ * 2);
  _Float16* sh2t   = (_Float16*)alloc((size_t)H_ * H_ * 2);
  _Float16* rh1t   = (_Float16*)alloc((size_t)H_ * KHD_ * 2);
  _Float16* rh2t   = (_Float16*)alloc((size_t)H_ * H_ * 2);

  auto gemm = [&](const _Float16* A, const _Float16* Bt, float* C, int M, int K, int Nc) {
    int waves  = (M / 16) * (Nc / 64);
    int blocks = (waves + 7) / 8;                 // 8 waves / 256-thread block
    k_gemm_wmma<<<blocks, 256, 0, stream>>>(A, Bt, C, M, K, Nc);
  };
  auto cvtT = [&](const float* W, _Float16* Wt, int K, int Ncols, int Kpad) {
    long long tot = (long long)Ncols * Kpad;
    k_cvt_transpose<<<NB(tot), 256, 0, stream>>>(W, Wt, tot, K, Ncols, Kpad);
  };

  // ---- graph structure
  k_fill_zero<<<NB(N_), 256, 0, stream>>>(deg, N_);
  k_fill_zero_i<<<1, 256, 0, stream>>>(cnts, B_);
  k_deg_count<<<NB(E_), 256, 0, stream>>>(dstI, deg, E_);
  k_batch_count<<<NB(N_), 256, 0, stream>>>(batch, cnts, N_);
  k_deg_finish<<<NB(N_), 256, 0, stream>>>(deg, dis, selfn, N_);
  k_scan_starts<<<1, 64, 0, stream>>>(cnts, starts, B_);

  // ---- precision conversions (weights transposed for WMMA B-fragment loads)
  k_cvt_pad<<<NB((long long)N_ * KP1_), 256, 0, stream>>>(x, xh, (long long)N_ * KP1_, EXT_, KP1_);
  cvtT(W1,   W1t,  EXT_, H_,  KP1_);
  cvtT(W2,   W2t,  H_,   H_,  H_);
  cvtT(W3,   W3t,  H_,   H_,  H_);
  cvtT(ph1w, ph1t, 2 + D_,        H3_, KHD_);
  cvtT(ph2w, ph2t, H3_,           H3_, H3_);
  cvtT(vh1w, vh1t, 2 + D_,        H_,  KHD_);
  cvtT(vh2w, vh2t, H_,            H_,  H_);
  cvtT(sh1w, sh1t, 2 + 4 + D_,    H_,  KHD_);
  cvtT(sh2w, sh2t, H_,            H_,  H_);
  cvtT(rh1w, rh1t, 2 + 4 + D_,    H_,  KHD_);
  cvtT(rh2w, rh2t, H_,            H_,  H_);

  // ---- 3 GCN layers + readouts
  auto gcn_layer = [&](const _Float16* hin_, int Kin, const _Float16* Wt,
                       const float* bias, _Float16* hout_, int roff) {
    long long nh = (long long)N_ * H_;
    gemm(hin_, Wt, tbuf, N_, Kin, H_);
    k_fill_zero<<<NB(nh), 256, 0, stream>>>(agg, nh);
    long long etot = (long long)E_ * (H_ / 4);
    k_edge_agg<<<NB(etot), 256, 0, stream>>>(srcI, dstI, tbuf, dis, agg, etot);
    k_gcn_finish<<<NB(nh), 256, 0, stream>>>(tbuf, agg, dis, selfn, bias, hout_, nh);
    k_readout<<<dim3(B_, H_ / 256), 256, 0, stream>>>(hout_, starts, cnts, comb, roff);
  };
  gcn_layer(xh, KP1_, W1t, b1, hA, 24);
  gcn_layer(hA, H_,   W2t, b2, hB, 24 + 1024);
  gcn_layer(hB, H_,   W3t, b3, hA, 24 + 2048);

  // ---- face
  k_face<<<B_, 256, 0, stream>>>(x, starts, cnts, comb);

  // ---- MLP heads
  auto head = [&](const float* p1, int n1, const float* p2, int n2,
                  const _Float16* w1t, const float* bias1, int h1,
                  const _Float16* w2t, const float* bias2,
                  const float* wo, const float* bo, int nout, float* outv) {
    k_build_head_in<<<NB((long long)B_ * KHD_), 256, 0, stream>>>(hin, p1, n1, p2, n2, comb);
    gemm(hin, w1t, a1, B_, KHD_, h1);
    long long tot = (long long)B_ * h1;
    k_bias_relu_cvt<<<NB(tot), 256, 0, stream>>>(a1, bias1, a1h, tot, h1);
    gemm(a1h, w2t, a2, B_, h1, h1);
    k_proj<<<NB(B_ * nout), 256, 0, stream>>>(a2, bias2, wo, bo, outv, h1, nout);
  };
  head(tpos, 2, nullptr, 0, ph1t, ph1b, H3_, ph2t, ph2b, posw, posb_, 2, posb);
  head(posb, 2, nullptr, 0, vh1t, vh1b, H_,  vh2t, vh2b, vecw, vecb_, 4, vecb);
  head(posb, 2, vecb,    4, sh1t, sh1b, H_,  sh2t, sh2b, scw,  scb_,  1, scb);
  head(posb, 2, vecb,    4, rh1t, rh1b, H_,  rh2t, rh2b, rotw, rotb_, 1, rotb);

  k_assemble<<<NB(B_ * 8), 256, 0, stream>>>(posb, scb, rotb, vecb, (float*)d_out);
}
